// SNN_CNN_Hybrid_1778116460710
// MI455X (gfx1250) — matrix operand order repaired and checked by reading, not compiled, not run on hardware
//
#include <hip/hip_runtime.h>
#include <math.h>

// ---------------------------------------------------------------------------
// SNN encoder (4-WGP pipelined sequential scan, spike bitmasks via ballot +
// TDM async staging) + WMMA im2col-GEMM decoder for gfx1250 (MI455X).
// Workspace requirement: ~120 MB.
// ---------------------------------------------------------------------------

typedef __attribute__((ext_vector_type(16))) _Float16 v16h;
typedef __attribute__((ext_vector_type(8)))  _Float16 h8;
typedef __attribute__((ext_vector_type(8)))  float    v8f;
typedef __attribute__((ext_vector_type(4)))  unsigned u32x4;
typedef __attribute__((ext_vector_type(8)))  int      i32x8;
typedef __attribute__((ext_vector_type(4)))  int      i32x4;

#define BETA 0.9f
#define NSTEPS 1000
#define SBIN 100

// Layer geometry
#define NP1 36800   // 4 x 50 x 184
#define NP2 18400   // 8 x 25 x 92
#define NP3 9568    // 16 x 13 x 46
#define NP4 5152    // 32 x 7 x 23
#define NB1 1150    // NP1/32
#define NB2 575
#define NB3 299

// ---------------------------------------------------------------------------
// Phase 1: conv1 pre-activations for all 1000 steps (massively parallel,
// memory bound). x: [1000,1,100,368] f32 -> c1pre: [1000,4,50,184] f16.
// ---------------------------------------------------------------------------
__global__ void k_conv1(const float* __restrict__ x, const float* __restrict__ w1,
                        const float* __restrict__ b1, _Float16* __restrict__ c1)
{
    long i = (long)blockIdx.x * blockDim.x + threadIdx.x;
    if (i >= (long)NSTEPS * NP1) return;
    int t = (int)(i / NP1);
    int p = (int)(i % NP1);
    int c = p / 9200, rem = p % 9200;
    int yy = rem / 184, xx = rem % 184;
    const float* xs = x + (long)t * 36800;
    __builtin_prefetch(xs + (2 * yy) * 368 + 2 * xx, 0, 1);  // global_prefetch_b8
    float acc = b1[c];
#pragma unroll
    for (int dy = 0; dy < 3; ++dy) {
        int iy = 2 * yy + dy;
        if (iy < 100) {
#pragma unroll
            for (int dx = 0; dx < 3; ++dx) {
                int ix = 2 * xx + dx;
                if (ix < 368) acc += w1[c * 9 + dy * 3 + dx] * xs[iy * 368 + ix];
            }
        }
    }
    c1[i] = (_Float16)acc;
}

// ---------------------------------------------------------------------------
// Phase 2 sync helpers: agent-scope acquire/release progress flags.
// ---------------------------------------------------------------------------
__device__ __forceinline__ void wait_flag(unsigned* flags, int idx, int t)
{
    if (threadIdx.x == 0) {
        unsigned spin = 0;
        while (__hip_atomic_load(&flags[idx], __ATOMIC_ACQUIRE,
                                 __HIP_MEMORY_SCOPE_AGENT) < (unsigned)(t + 1)) {
            __builtin_amdgcn_s_sleep(2);
            if (++spin > 2000000000u) break;  // safety bailout
        }
    }
    __syncthreads();
}

__device__ __forceinline__ void post_flag(unsigned* flags, int idx, int t)
{
    __syncthreads();
    if (threadIdx.x == 0) {
        __threadfence();
        __hip_atomic_store(&flags[idx], (unsigned)(t + 1), __ATOMIC_RELEASE,
                           __HIP_MEMORY_SCOPE_AGENT);
    }
}

// Stage NWORDS dwords global->LDS. Uses the Tensor Data Mover when available
// (one wave issues a 1-D D# descriptor; completion via s_wait_tensorcnt).
__device__ __forceinline__ void stage_bits(unsigned* s_bits, const unsigned* src,
                                           int nwords)
{
#if __has_builtin(__builtin_amdgcn_tensor_load_to_lds) && \
    __has_builtin(__builtin_amdgcn_s_wait_tensorcnt)
    if (threadIdx.x < 32) {
        // LDS byte offset = low 32 bits of the flat address (LDS aperture rule).
        unsigned ldsoff = (unsigned)(unsigned long long)(const void*)s_bits;
        unsigned long long ga = (unsigned long long)(const void*)src;
        u32x4 g0 = {0u, 0u, 0u, 0u};
        i32x8 g1 = {0, 0, 0, 0, 0, 0, 0, 0};
        i32x4 g2 = {0, 0, 0, 0};
        i32x4 g3 = {0, 0, 0, 0};
        i32x8 g4 = {0, 0, 0, 0, 0, 0, 0, 0};
        g0[0] = 1u;                                   // count=1, user descriptor
        g0[1] = ldsoff;                               // lds_addr
        g0[2] = (unsigned)ga;                         // global_addr[31:0]
        g0[3] = (unsigned)((ga >> 32) & 0x01FFFFFFull) | (2u << 30);  // [56:32] | type=2
        unsigned td0 = (unsigned)nwords;              // tensor dim0 (4B units)
        g1[0] = (int)(2u << 16);                      // data_size = 4 bytes
        g1[1] = (int)((td0 & 0xFFFFu) << 16);         // tensor_dim0[15:0]
        g1[2] = (int)(((td0 >> 16) & 0xFFFFu) | (1u << 16));  // dim0 hi | tensor_dim1=1
        g1[3] = (int)((td0 & 0xFFFFu) << 16);         // tile_dim0 = nwords
        g1[4] = 1;                                    // tile_dim1 = 1
        g1[5] = (int)td0;                             // tensor_dim0_stride
        __builtin_amdgcn_tensor_load_to_lds(g0, g1, g2, g3, g4, 0);
        __builtin_amdgcn_s_wait_tensorcnt((short)0);
    }
#else
    for (int i = threadIdx.x; i < nwords; i += 1024) s_bits[i] = src[i];
#endif
    __syncthreads();
}

// ---------------------------------------------------------------------------
// Layer 1: consumes precomputed c1pre; m in registers, sum/sq in LDS (294 KB).
// ---------------------------------------------------------------------------
template <int NE>
__device__ void scan_l1(const _Float16* __restrict__ c1pre, unsigned* __restrict__ outbits,
                        unsigned* flags, _Float16* __restrict__ skipdst, char* smem)
{
    const int tid = threadIdx.x;
    float* s_sum = (float*)smem;
    float* s_sq  = s_sum + NP1;
    float m[NE];
#pragma unroll
    for (int e = 0; e < NE; ++e) m[e] = 0.f;
    for (int i = tid; i < NP1; i += 1024) { s_sum[i] = 0.f; s_sq[i] = 0.f; }
    __syncthreads();

    for (int t = 0; t < NSTEPS; ++t) {
        const _Float16* cp = c1pre + (long)t * NP1;
#pragma unroll
        for (int e = 0; e < NE; ++e) {
            int p = e * 1024 + tid;
            bool valid = p < NP1;
            bool fire = false;
            if (valid) {
                float mm = BETA * m[e] + (float)cp[p];
                float u = mm - 1.0f;
                fire = u > 0.f;
                if (fire) mm -= 1.0f;
                m[e] = mm;
                s_sum[p] += mm;
                s_sq[p]  += mm * mm;
            }
            unsigned bal = __builtin_amdgcn_ballot_w32(fire);
            int p0 = e * 1024 + (tid & ~31);
            if ((tid & 31) == 0 && p0 < NP1) outbits[(long)t * NB1 + (p0 >> 5)] = bal;
        }
        post_flag(flags, 0, t);
        if ((t % SBIN) == SBIN - 1) {
            int k = t / SBIN;
#pragma unroll
            for (int e = 0; e < NE; ++e) {
                int p = e * 1024 + tid;
                if (p < NP1) {
                    int c = p / 9200, rem = p % 9200;
                    float mean = s_sum[p] * (1.0f / SBIN);
                    float var = s_sq[p] * (1.0f / SBIN) - mean * mean;
                    var = var > 1e-8f ? var : 1e-8f;
                    skipdst[(long)(16 + (k * 2 + 0) * 4 + c) * 9200 + rem] = (_Float16)mean;
                    skipdst[(long)(16 + (k * 2 + 1) * 4 + c) * 9200 + rem] = (_Float16)sqrtf(var);
                    s_sum[p] = 0.f; s_sq[p] = 0.f;
                }
            }
        }
    }
}

// ---------------------------------------------------------------------------
// Layers 2..4: bit-test conv from previous layer's spike bitmask (staged by TDM).
// ---------------------------------------------------------------------------
template <int NE, int Cout, int Cin, int H, int W, int Hin, int Win, int PT,
          int CHOFF, bool OUTB, int NBIN, int NBOUT>
__device__ void scan_layer(const float* __restrict__ wsrc, const float* __restrict__ bsrc,
                           const unsigned* __restrict__ inbits, unsigned* __restrict__ outbits,
                           unsigned* flags, int fin, int fout,
                           _Float16* __restrict__ skipdst, char* smem)
{
    const int NP = Cout * H * W;
    const int tid = threadIdx.x;
    float* s_m   = (float*)smem;
    float* s_sum = s_m + NP;
    float* s_sq  = s_sum + NP;
    float* s_w   = s_sq + NP;
    float* s_b   = s_w + Cout * Cin * 9;
    unsigned* s_bits = (unsigned*)(s_b + Cout);

    for (int i = tid; i < NP; i += 1024) { s_m[i] = 0.f; s_sum[i] = 0.f; s_sq[i] = 0.f; }
    for (int i = tid; i < Cout * Cin * 9; i += 1024) s_w[i] = wsrc[i];
    for (int i = tid; i < Cout; i += 1024) s_b[i] = bsrc[i];
    __syncthreads();

    for (int t = 0; t < NSTEPS; ++t) {
        wait_flag(flags, fin, t);
        __threadfence();
        stage_bits(s_bits, inbits + (long)t * NBIN, NBIN);

        for (int e = 0; e < NE; ++e) {
            int p = e * 1024 + tid;
            bool valid = p < NP;
            bool fire = false;
            if (valid) {
                int c = p / (H * W), rem = p % (H * W);
                int y = rem / W, x = rem % W;
                float acc = s_b[c];
                for (int ci = 0; ci < Cin; ++ci) {
                    int wb = (c * Cin + ci) * 9;
                    int ib = ci * (Hin * Win);
#pragma unroll
                    for (int dy = 0; dy < 3; ++dy) {
                        int iy = 2 * y + dy - PT;
                        if (iy >= 0 && iy < Hin) {
                            int rb = ib + iy * Win + 2 * x;
#pragma unroll
                            for (int dx = 0; dx < 3; ++dx) {
                                int ix = 2 * x + dx;
                                if (ix < Win) {
                                    int bi = rb + dx;
                                    if ((s_bits[bi >> 5] >> (bi & 31)) & 1u)
                                        acc += s_w[wb + dy * 3 + dx];
                                }
                            }
                        }
                    }
                }
                float mm = BETA * s_m[p] + acc;
                float u = mm - 1.0f;
                fire = u > 0.f;
                if (fire) mm -= 1.0f;
                s_m[p] = mm;
                s_sum[p] += mm;
                s_sq[p]  += mm * mm;
            }
            if (OUTB && outbits) {
                unsigned bal = __builtin_amdgcn_ballot_w32(fire);
                int p0 = e * 1024 + (tid & ~31);
                if ((tid & 31) == 0 && p0 < NP) outbits[(long)t * NBOUT + (p0 >> 5)] = bal;
            }
        }
        post_flag(flags, fout, t);
        if ((t % SBIN) == SBIN - 1) {
            int k = t / SBIN;
            for (int e = 0; e < NE; ++e) {
                int p = e * 1024 + tid;
                if (p < NP) {
                    int c = p / (H * W), rem = p % (H * W);
                    float mean = s_sum[p] * (1.0f / SBIN);
                    float var = s_sq[p] * (1.0f / SBIN) - mean * mean;
                    var = var > 1e-8f ? var : 1e-8f;
                    skipdst[(long)(CHOFF + (k * 2 + 0) * Cout + c) * (H * W) + rem] = (_Float16)mean;
                    skipdst[(long)(CHOFF + (k * 2 + 1) * Cout + c) * (H * W) + rem] = (_Float16)sqrtf(var);
                    s_sum[p] = 0.f; s_sq[p] = 0.f;
                }
            }
        }
    }
}

__global__ void __launch_bounds__(1024, 1)
k_snn_scan(const _Float16* c1pre,
           const float* w2, const float* b2, const float* w3, const float* b3,
           const float* w4, const float* b4,
           unsigned* spk1b, unsigned* spk2b, unsigned* spk3b, unsigned* flags,
           _Float16* in1buf, _Float16* in2buf, _Float16* in3buf, _Float16* in4buf)
{
    extern __shared__ char smem[];
    switch (blockIdx.x) {
    case 0: scan_l1<36>(c1pre, spk1b, flags, in1buf, smem); break;
    case 1: scan_layer<18, 8, 4, 25, 92, 50, 184, 0, 32, true, NB1, NB2>(
                w2, b2, spk1b, spk2b, flags, 0, 1, in2buf, smem); break;
    case 2: scan_layer<10, 16, 8, 13, 46, 25, 92, 1, 64, true, NB2, NB3>(
                w3, b3, spk2b, spk3b, flags, 1, 2, in3buf, smem); break;
    case 3: scan_layer<6, 32, 16, 7, 23, 13, 46, 1, 0, false, NB3, 1>(
                w4, b4, spk3b, nullptr, flags, 2, 3, in4buf, smem); break;
    }
}

// ---------------------------------------------------------------------------
// Decoder, step A: explicit im2col (f16) with M padded to 32 so the GEMM
// K-loop is branch-free. A2: [Mp][Ktot], row-major, 16B-aligned rows.
// ---------------------------------------------------------------------------
__global__ void k_im2col(const _Float16* __restrict__ in, _Float16* __restrict__ A2,
                         int Cin, int H, int W, int Mp)
{
    int Ktot = Cin * 9;
    long i = (long)blockIdx.x * blockDim.x + threadIdx.x;
    long total = (long)Mp * Ktot;
    if (i >= total) return;
    int mp = (int)(i / Ktot);
    int k  = (int)(i % Ktot);
    int M = H * W;
    _Float16 v = (_Float16)0.f;
    if (mp < M) {
        int ci = k / 9, r = k % 9;
        int y = mp / W, x = mp % W;
        int iy = y + r / 3 - 1, ix = x + r % 3 - 1;
        if (iy >= 0 && iy < H && ix >= 0 && ix < W) v = in[ci * M + iy * W + ix];
    }
    A2[i] = v;
}

// Decoder, step B: weights -> f16, Cout padded to Np (zero rows).
__global__ void k_wcvt(const float* __restrict__ w, _Float16* __restrict__ wf,
                       int Cout, int Ktot, int Np)
{
    long i = (long)blockIdx.x * blockDim.x + threadIdx.x;
    long total = (long)Np * Ktot;
    if (i >= total) return;
    int n = (int)(i / Ktot);
    wf[i] = (n < Cout) ? (_Float16)w[i] : (_Float16)0.f;
}

// ---------------------------------------------------------------------------
// Decoder, step C: dense GEMM out[Cout][M] = relu(A2[Mp][K] x Wf[Np][K]^T + b)
// on v_wmma_f32_16x16x32_f16. Per wave: 2 M-tiles x 1 N-tile sharing the B
// fragment. Fragment gathers are contiguous 128-bit loads:
//   A lane: K runs {b..b+7} and {b+16..b+23}, b = 8*(lane>=16)
//   B lane: same K runs as A
//   D lane: N = lane&15, M = r + 8*(lane>=16)
// ---------------------------------------------------------------------------
__global__ void __launch_bounds__(256)
k_wmma_gemm(const _Float16* __restrict__ A2, const _Float16* __restrict__ Wf,
            const float* __restrict__ bias, _Float16* __restrict__ out,
            int M, int Mp, int Cout, int Ktot)
{
    const int lane = threadIdx.x & 31;
    const int wv = threadIdx.x >> 5;
    const int Mt2 = Mp >> 5;                 // 32-row supertiles
    const int Ntiles = (Cout + 15) >> 4;
    const int g = blockIdx.x * (blockDim.x >> 5) + wv;
    if (g >= Mt2 * Ntiles) return;
    const int mt = g % Mt2, nt = g / Mt2;
    const int lm = lane & 15;
    const int hi = (lane >> 4) & 1;

    const h8* __restrict__ pa0 = (const h8*)(A2 + (size_t)(mt * 32 + lm) * Ktot);
    const h8* __restrict__ pa1 = (const h8*)(A2 + (size_t)(mt * 32 + 16 + lm) * Ktot);
    const h8* __restrict__ pb  = (const h8*)(Wf + (size_t)(nt * 16 + lm) * Ktot);
    const int ao = hi;       // +8 halfs for upper half-wave
    const int bo = hi;       // +8 halfs for upper half-wave

    v8f acc0 = {}, acc1 = {};
    const int kv = Ktot >> 3;  // row length in h8 units (Ktot % 32 == 0 always)
#pragma unroll 2
    for (int ki = 0; ki < kv; ki += 4) {
        h8 b0 = pb[ki + bo], b1 = pb[ki + bo + 2];
        v16h bf = __builtin_shufflevector(b0, b1, 0, 1, 2, 3, 4, 5, 6, 7,
                                          8, 9, 10, 11, 12, 13, 14, 15);
        h8 a0l = pa0[ki + ao], a0h = pa0[ki + ao + 2];
        v16h af0 = __builtin_shufflevector(a0l, a0h, 0, 1, 2, 3, 4, 5, 6, 7,
                                           8, 9, 10, 11, 12, 13, 14, 15);
        acc0 = __builtin_amdgcn_wmma_f32_16x16x32_f16(false, af0, false, bf,
                                                      (short)0, acc0, false, false);
        h8 a1l = pa1[ki + ao], a1h = pa1[ki + ao + 2];
        v16h af1 = __builtin_shufflevector(a1l, a1h, 0, 1, 2, 3, 4, 5, 6, 7,
                                           8, 9, 10, 11, 12, 13, 14, 15);
        acc1 = __builtin_amdgcn_wmma_f32_16x16x32_f16(false, af1, false, bf,
                                                      (short)0, acc1, false, false);
    }

    const int n = nt * 16 + lm;
    if (n < Cout) {
        float bv = bias[n];
#pragma unroll
        for (int r = 0; r < 8; ++r) {
            int pr0 = mt * 32 + r + hi * 8;
            if (pr0 < M) {
                float v = acc0[r] + bv;
                out[(size_t)n * M + pr0] = (_Float16)(v > 0.f ? v : 0.f);
            }
            int pr1 = mt * 32 + 16 + r + hi * 8;
            if (pr1 < M) {
                float v = acc1[r] + bv;
                out[(size_t)n * M + pr1] = (_Float16)(v > 0.f ? v : 0.f);
            }
        }
    }
}

// ---------------------------------------------------------------------------
// Bilinear upsample (half-pixel centers, edge clamp); writes channels [0,C)
// of the destination buffer (channel stride H*W).
// ---------------------------------------------------------------------------
__global__ void k_upsample(const _Float16* __restrict__ in, _Float16* __restrict__ dst,
                           int C, int h, int w, int H, int W)
{
    int i = blockIdx.x * blockDim.x + threadIdx.x;
    int total = C * H * W;
    if (i >= total) return;
    int c = i / (H * W), rem = i % (H * W);
    int Y = rem / W, X = rem % W;
    float fy = ((Y + 0.5f) * h) / H - 0.5f;
    float fx = ((X + 0.5f) * w) / W - 0.5f;
    fy = fy < 0.f ? 0.f : fy;
    fx = fx < 0.f ? 0.f : fx;
    int y0 = (int)fy, x0 = (int)fx;
    int y1 = y0 + 1 < h ? y0 + 1 : h - 1;
    int x1 = x0 + 1 < w ? x0 + 1 : w - 1;
    float ty = fy - y0, tx = fx - x0;
    const _Float16* ip = in + c * h * w;
    float v00 = (float)ip[y0 * w + x0], v01 = (float)ip[y0 * w + x1];
    float v10 = (float)ip[y1 * w + x0], v11 = (float)ip[y1 * w + x1];
    float v = (1.f - ty) * ((1.f - tx) * v00 + tx * v01) +
              ty * ((1.f - tx) * v10 + tx * v11);
    dst[c * H * W + rem] = (_Float16)v;
}

// Final: upsample d1 (8ch,50x184 -> 100x368) fused with 1x1 conv -> f32 out.
__global__ void k_output(const _Float16* __restrict__ d1, const float* __restrict__ wo,
                         const float* __restrict__ bo, float* __restrict__ out)
{
    int i = blockIdx.x * blockDim.x + threadIdx.x;
    if (i >= 36800) return;
    int Y = i / 368, X = i % 368;
    float fy = ((Y + 0.5f) * 50.f) / 100.f - 0.5f;
    float fx = ((X + 0.5f) * 184.f) / 368.f - 0.5f;
    fy = fy < 0.f ? 0.f : fy;
    fx = fx < 0.f ? 0.f : fx;
    int y0 = (int)fy, x0 = (int)fx;
    int y1 = y0 + 1 < 50 ? y0 + 1 : 49;
    int x1 = x0 + 1 < 184 ? x0 + 1 : 183;
    float ty = fy - y0, tx = fx - x0;
    float acc = bo[0];
#pragma unroll
    for (int c = 0; c < 8; ++c) {
        const _Float16* ip = d1 + c * 9200;
        float v00 = (float)ip[y0 * 184 + x0], v01 = (float)ip[y0 * 184 + x1];
        float v10 = (float)ip[y1 * 184 + x0], v11 = (float)ip[y1 * 184 + x1];
        acc += wo[c] * ((1.f - ty) * ((1.f - tx) * v00 + tx * v01) +
                        ty * ((1.f - tx) * v10 + tx * v11));
    }
    out[i] = acc;
}

// ---------------------------------------------------------------------------
extern "C" void kernel_launch(void* const* d_in, const int* in_sizes, int n_in,
                              void* d_out, int out_size, void* d_ws, size_t ws_size,
                              hipStream_t stream)
{
    (void)in_sizes; (void)n_in; (void)out_size; (void)ws_size;
    const float* x   = (const float*)d_in[0];
    const float* w1  = (const float*)d_in[1];  const float* b1  = (const float*)d_in[2];
    const float* w2  = (const float*)d_in[3];  const float* b2  = (const float*)d_in[4];
    const float* w3  = (const float*)d_in[5];  const float* b3  = (const float*)d_in[6];
    const float* w4  = (const float*)d_in[7];  const float* b4  = (const float*)d_in[8];
    const float* dw4 = (const float*)d_in[9];  const float* db4 = (const float*)d_in[10];
    const float* dw3 = (const float*)d_in[11]; const float* db3 = (const float*)d_in[12];
    const float* dw2 = (const float*)d_in[13]; const float* db2 = (const float*)d_in[14];
    const float* dw1 = (const float*)d_in[15]; const float* db1 = (const float*)d_in[16];
    const float* wo  = (const float*)d_in[17]; const float* bo  = (const float*)d_in[18];

    char* ws = (char*)d_ws;
    size_t off = 0;
    auto walloc = [&](size_t bytes) -> void* {
        void* p = ws + off;
        off = (off + bytes + 255) & ~(size_t)255;
        return p;
    };
    _Float16* c1pre  = (_Float16*)walloc((size_t)NSTEPS * NP1 * 2);
    unsigned* spk1b  = (unsigned*)walloc((size_t)NSTEPS * NB1 * 4);
    unsigned* spk2b  = (unsigned*)walloc((size_t)NSTEPS * NB2 * 4);
    unsigned* spk3b  = (unsigned*)walloc((size_t)NSTEPS * NB3 * 4);
    unsigned* flags  = (unsigned*)walloc(256);
    _Float16* in4buf = (_Float16*)walloc((size_t)640 * 161 * 2);
    _Float16* in3buf = (_Float16*)walloc((size_t)384 * 598 * 2);
    _Float16* in2buf = (_Float16*)walloc((size_t)192 * 2300 * 2);
    _Float16* in1buf = (_Float16*)walloc((size_t)96 * 9200 * 2);
    _Float16* d4     = (_Float16*)walloc((size_t)64 * 161 * 2);
    _Float16* d3     = (_Float16*)walloc((size_t)32 * 598 * 2);
    _Float16* d2     = (_Float16*)walloc((size_t)16 * 2300 * 2);
    _Float16* d1     = (_Float16*)walloc((size_t)8 * 9200 * 2);
    // im2col + f16 weight buffers (M padded to 32, Cout padded to 16)
    _Float16* A24 = (_Float16*)walloc((size_t)192 * 5760 * 2);
    _Float16* A23 = (_Float16*)walloc((size_t)608 * 3456 * 2);
    _Float16* A22 = (_Float16*)walloc((size_t)2304 * 1728 * 2);
    _Float16* A21 = (_Float16*)walloc((size_t)9216 * 864 * 2);
    _Float16* Wf4 = (_Float16*)walloc((size_t)64 * 5760 * 2);
    _Float16* Wf3 = (_Float16*)walloc((size_t)32 * 3456 * 2);
    _Float16* Wf2 = (_Float16*)walloc((size_t)16 * 1728 * 2);
    _Float16* Wf1 = (_Float16*)walloc((size_t)16 * 864 * 2);

    (void)hipMemsetAsync(flags, 0, 64, stream);

    // Phase 1: conv1 pre-activations, whole chip, memory-bound.
    {
        long n = (long)NSTEPS * NP1;
        k_conv1<<<(unsigned)((n + 255) / 256), 256, 0, stream>>>(x, w1, b1, c1pre);
    }

    // Phase 2: 4-WGP pipelined sequential scan. Layer1 needs 2*NP1*4 = 294400 B LDS.
    const int SMEM = 2 * NP1 * 4;
    (void)hipFuncSetAttribute((const void*)k_snn_scan,
                              hipFuncAttributeMaxDynamicSharedMemorySize, SMEM);
    k_snn_scan<<<4, 1024, SMEM, stream>>>(c1pre, w2, b2, w3, b3, w4, b4,
                                          spk1b, spk2b, spk3b, flags,
                                          in1buf, in2buf, in3buf, in4buf);

    // Phase 3: WMMA decoder (im2col -> dense f16 GEMM -> upsample chain).
    auto conv = [&](const _Float16* in, _Float16* A2, const float* w, _Float16* Wf,
                    const float* b, _Float16* outp,
                    int Cin, int Cout, int H, int W) {
        int M = H * W;
        int Mp = (M + 31) & ~31;
        int Np = (Cout + 15) & ~15;
        int Ktot = Cin * 9;
        long ni = (long)Mp * Ktot;
        k_im2col<<<(unsigned)((ni + 255) / 256), 256, 0, stream>>>(in, A2, Cin, H, W, Mp);
        long nw = (long)Np * Ktot;
        k_wcvt<<<(unsigned)((nw + 255) / 256), 256, 0, stream>>>(w, Wf, Cout, Ktot, Np);
        int tiles = (Mp / 32) * (Np / 16);
        k_wmma_gemm<<<(tiles + 7) / 8, 256, 0, stream>>>(A2, Wf, b, outp, M, Mp, Cout, Ktot);
    };
    conv(in4buf, A24, dw4, Wf4, db4, d4, 640, 64, 7, 23);
    k_upsample<<<(64 * 13 * 46 + 255) / 256, 256, 0, stream>>>(d4, in3buf, 64, 7, 23, 13, 46);
    conv(in3buf, A23, dw3, Wf3, db3, d3, 384, 32, 13, 46);
    k_upsample<<<(32 * 25 * 92 + 255) / 256, 256, 0, stream>>>(d3, in2buf, 32, 13, 46, 25, 92);
    conv(in2buf, A22, dw2, Wf2, db2, d2, 192, 16, 25, 92);
    k_upsample<<<(16 * 50 * 184 + 255) / 256, 256, 0, stream>>>(d2, in1buf, 16, 25, 92, 50, 184);
    conv(in1buf, A21, dw1, Wf1, db1, d1, 96, 8, 50, 184);
    k_output<<<(36800 + 255) / 256, 256, 0, stream>>>(d1, wo, bo, (float*)d_out);
}
